// SciNO_probe_11725260718535
// MI455X (gfx1250) — matrix operand order repaired and checked by reading, not compiled, not run on hardware
//
#include <hip/hip_runtime.h>

#define B_DIM 512
#define N_DIM 512
#define H_DIM 2560
#define H2    5120
#define S_DIM 1536
#define F_DIM 128
#define L_DIM 4

typedef __attribute__((ext_vector_type(16))) __bf16 v16bf;
typedef __attribute__((ext_vector_type(8)))  float  v8f;

__device__ __forceinline__ unsigned f2bf(float f) {
    unsigned u = __float_as_uint(f);
    u += 0x7FFFu + ((u >> 16) & 1u);   // round-to-nearest-even
    return u >> 16;
}

union Frag { v16bf v; uint2 d[4]; };

constexpr int EPI_BIAS  = 0;  // out = acc + bias[n]
constexpr int EPI_LEAKY = 1;  // out = leaky(acc + bias[n])
constexpr int EPI_TEMB  = 2;  // out = acc * aux[row*(N/2) + (col>>1)]
constexpr int EPI_SKIP  = 3;  // out = acc + aux[row*N + col]

constexpr int LDR = 36;       // LDS row stride in shorts (72B: bank stride 18 -> conflict-free)

// C[M,N] = A[M,K] @ B[K,N] (+ epilogue).
// A: fp32 row-major or bf16 row-major. B: fp32 row-major [K][N] or bf16 COLUMN-major ([N][K] rows).
// Block tile 128x128, BK=32, 8 waves each computing 64x32 (8 WMMA tiles), double-buffered LDS.
// M%128==0, N%128==0, K%32==0 for every GEMM here.
template<int EPI, bool A_BF16, bool B_BF16CM, bool OUT_BF16>
__global__ __launch_bounds__(256) void gemm_wmma(
    const void* __restrict__ Av, const void* __restrict__ Bv,
    const float* __restrict__ bias, const float* __restrict__ aux,
    void* __restrict__ Cv, int M, int N, int K)
{
    __shared__ alignas(16) unsigned short ldsA[2][128 * LDR];
    __shared__ alignas(16) unsigned short ldsB[2][128 * LDR];

    const int tid     = threadIdx.x;
    const int lane    = tid & 31;
    const int wid     = tid >> 5;
    const int rowBase = blockIdx.y * 128;
    const int colBase = blockIdx.x * 128;
    const int wm      = (wid >> 2) * 64;   // 2 waves along M
    const int wn      = (wid & 3) * 32;    // 4 waves along N
    const int l15     = lane & 15;
    const int lhi     = lane >> 4;
    const int ac2     = lhi * 2;           // A K-chunk uint2 index (koff = lhi*8)
    const int bc4     = lhi * 4;           // unused: B reads use ac2, same k order as A

    const v8f vzero = {0.f, 0.f, 0.f, 0.f, 0.f, 0.f, 0.f, 0.f};
    v8f acc[4][2];
    #pragma unroll
    for (int i = 0; i < 4; i++)
        #pragma unroll
        for (int j = 0; j < 2; j++) acc[i][j] = vzero;

    auto stageA = [&](int buf, int k0) {
        if constexpr (A_BF16) {
            const unsigned short* Ag = (const unsigned short*)Av;   // [M][K] bf16
            #pragma unroll
            for (int i = 0; i < 2; i++) {
                int idx = tid + i * 256;              // 512 quads: 128 rows x 4
                int r = idx >> 2, q = (idx & 3) * 8;
                uint4 v = *(const uint4*)(Ag + (size_t)(rowBase + r) * K + k0 + q);
                *(uint2*)&ldsA[buf][r * LDR + q]     = make_uint2(v.x, v.y);
                *(uint2*)&ldsA[buf][r * LDR + q + 4] = make_uint2(v.z, v.w);
            }
        } else {
            const float* Ag = (const float*)Av;       // [M][K] fp32
            #pragma unroll
            for (int i = 0; i < 4; i++) {
                int idx = tid + i * 256;              // 1024 float4s: 128 rows x 8
                int r = idx >> 3, c4 = (idx & 7) * 4;
                float4 f = *(const float4*)(Ag + (size_t)(rowBase + r) * K + k0 + c4);
                unsigned lo = f2bf(f.x) | (f2bf(f.y) << 16);
                unsigned hi = f2bf(f.z) | (f2bf(f.w) << 16);
                *(uint2*)&ldsA[buf][r * LDR + c4] = make_uint2(lo, hi);
            }
        }
    };
    auto stageB = [&](int buf, int k0) {
        if constexpr (B_BF16CM) {
            const unsigned short* Bg = (const unsigned short*)Bv;   // [N][K] bf16
            #pragma unroll
            for (int i = 0; i < 2; i++) {
                int idx = tid + i * 256;
                int n = idx >> 2, q = (idx & 3) * 8;
                uint4 v = *(const uint4*)(Bg + (size_t)(colBase + n) * K + k0 + q);
                *(uint2*)&ldsB[buf][n * LDR + q]     = make_uint2(v.x, v.y);
                *(uint2*)&ldsB[buf][n * LDR + q + 4] = make_uint2(v.z, v.w);
            }
        } else {
            const float* Bg = (const float*)Bv;       // [K][N] fp32 -> transpose into [n][k]
            #pragma unroll
            for (int i = 0; i < 4; i++) {
                int idx = tid + i * 256;              // 1024 float4s: 32 k-rows x 32
                int kr = idx >> 5, c4 = (idx & 31) * 4;
                float4 f = *(const float4*)(Bg + (size_t)(k0 + kr) * N + colBase + c4);
                ldsB[buf][(c4 + 0) * LDR + kr] = (unsigned short)f2bf(f.x);
                ldsB[buf][(c4 + 1) * LDR + kr] = (unsigned short)f2bf(f.y);
                ldsB[buf][(c4 + 2) * LDR + kr] = (unsigned short)f2bf(f.z);
                ldsB[buf][(c4 + 3) * LDR + kr] = (unsigned short)f2bf(f.w);
            }
        }
    };

    stageA(0, 0);
    stageB(0, 0);

    int buf = 0;
    for (int k0 = 0; k0 < K; k0 += 32) {
        __syncthreads();
        if (k0 + 32 < K) { stageA(buf ^ 1, k0 + 32); stageB(buf ^ 1, k0 + 32); }
        if (k0 + 64 < K) {              // prefetch two tiles ahead -> global_prefetch
            int r = tid >> 1, h = (tid & 1) * 16;
            if constexpr (A_BF16)
                __builtin_prefetch((const unsigned short*)Av + (size_t)(rowBase + r) * K + k0 + 64 + h, 0, 1);
            else
                __builtin_prefetch((const float*)Av + (size_t)(rowBase + r) * K + k0 + 64 + h, 0, 1);
            if constexpr (B_BF16CM)
                __builtin_prefetch((const unsigned short*)Bv + (size_t)(colBase + r) * K + k0 + 64 + h, 0, 1);
            else {
                int kr = tid >> 3, c = (tid & 7) * 16;
                __builtin_prefetch((const float*)Bv + (size_t)(k0 + 64 + kr) * N + colBase + c, 0, 1);
            }
        }

        Frag a[4], b[2];
        #pragma unroll
        for (int tm = 0; tm < 4; tm++) {
            const uint2* rp = (const uint2*)&ldsA[buf][(wm + tm * 16 + l15) * LDR];
            a[tm].d[0] = rp[ac2];     a[tm].d[1] = rp[ac2 + 1];   // K = koff..koff+7
            a[tm].d[2] = rp[ac2 + 4]; a[tm].d[3] = rp[ac2 + 5];   // K = 16+koff..
        }
        #pragma unroll
        for (int tn = 0; tn < 2; tn++) {
            const uint2* rp = (const uint2*)&ldsB[buf][(wn + tn * 16 + l15) * LDR];
            a:;
            b[tn].d[0] = rp[ac2];     b[tn].d[1] = rp[ac2 + 1];
            b[tn].d[2] = rp[ac2 + 4]; b[tn].d[3] = rp[ac2 + 5];
        }
        #pragma unroll
        for (int tm = 0; tm < 4; tm++)
            #pragma unroll
            for (int tn = 0; tn < 2; tn++)
                acc[tm][tn] = __builtin_amdgcn_wmma_f32_16x16x32_bf16(
                    false, a[tm].v, false, b[tn].v, (short)0, acc[tm][tn],
                    false, false);
        buf ^= 1;
    }

    // epilogue: C/D layout: n = lane&15, m = (lane>>4)*8 + vgpr_index
    #pragma unroll
    for (int tm = 0; tm < 4; tm++) {
        #pragma unroll
        for (int tn = 0; tn < 2; tn++) {
            const int col = colBase + wn + tn * 16 + l15;
            const int r0  = rowBase + wm + tm * 16 + lhi * 8;
            float bv = 0.f;
            if (EPI == EPI_BIAS || EPI == EPI_LEAKY) bv = bias[col];
            #pragma unroll
            for (int r = 0; r < 8; r++) {
                float v = acc[tm][tn][r];
                const int row = r0 + r;
                if (EPI == EPI_BIAS)       v += bv;
                else if (EPI == EPI_LEAKY) { v += bv; v = v >= 0.f ? v : 0.01f * v; }
                else if (EPI == EPI_TEMB)  v *= aux[(size_t)row * (N >> 1) + (col >> 1)];
                else if (EPI == EPI_SKIP)  v += aux[(size_t)row * N + col];
                if constexpr (OUT_BF16)
                    ((unsigned short*)Cv)[(size_t)row * N + col] = (unsigned short)f2bf(v);
                else
                    ((float*)Cv)[(size_t)row * N + col] = v;
            }
        }
    }
}

// csum[h] = sum_n W_te[n,h]  (t2 @ W_te degenerates to t[b]*csum[h])
__global__ void k_colsum(const float* __restrict__ W, float* __restrict__ out) {
    int h = blockIdx.x * blockDim.x + threadIdx.x;
    if (h >= H_DIM) return;
    float s = 0.f;
    for (int n = 0; n < N_DIM; n++) s += W[(size_t)n * H_DIM + h];
    out[h] = s;
}

__global__ void k_temb(const float* __restrict__ t, const float* __restrict__ csum,
                       const float* __restrict__ bte, float* __restrict__ temb) {
    int id = blockIdx.x * blockDim.x + threadIdx.x;   // B*H
    int b = id / H_DIM, h = id - b * H_DIM;
    temb[id] = sinf(t[b] * csum[h] + bte[h]);
}

// E[n,2k]=cos(2pi nk/H)/sqrt(H), E[n,2k+1]=-sin(...)/sqrt(H); D = E^T. bf16.
// Note: D's array doubles as column-major E, and E's array as column-major D.
__global__ void k_tables(unsigned short* __restrict__ E, unsigned short* __restrict__ D) {
    int id = blockIdx.x * blockDim.x + threadIdx.x;   // H*H
    int n = id / H_DIM, k = id - n * H_DIM;
    int m = (n * k) % H_DIM;                          // exact integer phase
    float ang = (float)m * (6.2831853071795864f / (float)H_DIM);
    float s, c;
    sincosf(ang, &s, &c);
    const float inv = 0.019764235376052372f;          // 1/sqrt(2560)
    unsigned short bc = (unsigned short)f2bf(c * inv);
    unsigned short bs = (unsigned short)f2bf(-s * inv);
    *(unsigned*)&E[(size_t)n * H2 + 2 * k] = (unsigned)bc | ((unsigned)bs << 16);
    D[(size_t)(2 * k) * H_DIM + n]     = bc;
    D[(size_t)(2 * k + 1) * H_DIM + n] = bs;
}

__global__ void k_gather(const float* __restrict__ full, const int* __restrict__ idx,
                         float* __restrict__ out) {
    int id = blockIdx.x * blockDim.x + threadIdx.x;   // B*F
    int b = id >> 7, f = id & (F_DIM - 1);
    out[id] = full[(size_t)b * N_DIM + idx[f]];
}

extern "C" void kernel_launch(void* const* d_in, const int* in_sizes, int n_in,
                              void* d_out, int out_size, void* d_ws, size_t ws_size,
                              hipStream_t stream) {
    const float* x   = (const float*)d_in[0];
    const float* t   = (const float*)d_in[1];
    const int*   fn  = (const int*)d_in[2];
    const float* Wte = (const float*)d_in[3];
    const float* bte = (const float*)d_in[4];
    const float* W1  = (const float*)d_in[5];
    const float* b1  = (const float*)d_in[6];
    const float* Wf  = (const float*)d_in[7];
    const float* bfv = (const float*)d_in[8];
    const float* Wp1 = (const float*)d_in[9];
    const float* bp1 = (const float*)d_in[10];
    const float* Wp2 = (const float*)d_in[11];
    const float* bp2 = (const float*)d_in[12];
    const float* Wp3 = (const float*)d_in[13];
    const float* bp3 = (const float*)d_in[14];
    float* out = (float*)d_out;

    char* p = (char*)d_ws;
    auto carve = [&](size_t bytes) { char* r = p; p += (bytes + 255) & ~(size_t)255; return r; };
    float*          csum = (float*)carve((size_t)H_DIM * 4);
    float*          temb = (float*)carve((size_t)B_DIM * H_DIM * 4);
    unsigned short* E    = (unsigned short*)carve((size_t)H_DIM * H2 * 2);
    unsigned short* D    = (unsigned short*)carve((size_t)H2 * H_DIM * 2);
    float*          X0   = (float*)carve((size_t)B_DIM * H_DIM * 4);
    float*          X1   = (float*)carve((size_t)B_DIM * H_DIM * 4);
    unsigned short* comb = (unsigned short*)carve((size_t)B_DIM * H2 * 2);
    unsigned short* Xt   = (unsigned short*)carve((size_t)B_DIM * H2 * 2);
    unsigned short* h1   = (unsigned short*)carve((size_t)B_DIM * H_DIM * 2);
    unsigned short* h2   = (unsigned short*)carve((size_t)B_DIM * S_DIM * 2);
    float*          full = (float*)carve((size_t)B_DIM * N_DIM * 4);

    k_colsum<<<(H_DIM + 255) / 256, 256, 0, stream>>>(Wte, csum);
    k_temb<<<(B_DIM * H_DIM) / 256, 256, 0, stream>>>(t, csum, bte, temb);
    k_tables<<<(H_DIM * H_DIM) / 256, 256, 0, stream>>>(E, D);

    dim3 blk(256);
    auto grid = [](int Nn) { return dim3(Nn / 128, B_DIM / 128); };

    // X0 = x @ W1 + b1
    gemm_wmma<EPI_BIAS, false, false, false><<<grid(H_DIM), blk, 0, stream>>>(
        x, W1, b1, nullptr, X0, B_DIM, H_DIM, N_DIM);

    float* Xc = X0; float* Xn = X1;
    for (int i = 0; i < L_DIM; i++) {
        // comb = (Xc @ E) * temb; B = E column-major == D's array
        gemm_wmma<EPI_TEMB, false, true, true><<<grid(H2), blk, 0, stream>>>(
            Xc, D, nullptr, temb, comb, B_DIM, H2, H_DIM);
        // Xt = comb @ Wf[i] + bf[i]
        gemm_wmma<EPI_BIAS, true, false, true><<<grid(H2), blk, 0, stream>>>(
            comb, Wf + (size_t)i * H2 * H2, bfv + (size_t)i * H2, nullptr,
            Xt, B_DIM, H2, H2);
        // Xn = Xt @ D + Xc; B = D column-major == E's array
        gemm_wmma<EPI_SKIP, true, true, false><<<grid(H_DIM), blk, 0, stream>>>(
            Xt, E, nullptr, Xc, Xn, B_DIM, H_DIM, H2);
        float* tmp = Xc; Xc = Xn; Xn = tmp;
    }

    gemm_wmma<EPI_LEAKY, false, false, true><<<grid(H_DIM), blk, 0, stream>>>(
        Xc, Wp1, bp1, nullptr, h1, B_DIM, H_DIM, H_DIM);
    gemm_wmma<EPI_LEAKY, true, false, true><<<grid(S_DIM), blk, 0, stream>>>(
        h1, Wp2, bp2, nullptr, h2, B_DIM, S_DIM, H_DIM);
    gemm_wmma<EPI_BIAS, true, false, false><<<grid(N_DIM), blk, 0, stream>>>(
        h2, Wp3, bp3, nullptr, full, B_DIM, N_DIM, S_DIM);

    k_gather<<<(B_DIM * F_DIM) / 256, 256, 0, stream>>>(full, fn, out);

    (void)in_sizes; (void)n_in; (void)out_size; (void)ws_size;
}
